// HybridLinearAttention_12395275616533
// MI455X (gfx1250) — matrix operand order repaired and checked by reading, not compiled, not run on hardware
//
#include <hip/hip_runtime.h>
#include <hip/hip_bf16.h>

typedef __bf16 bf16;
typedef __attribute__((ext_vector_type(16))) __bf16 v16bf;
typedef __attribute__((ext_vector_type(8)))  float  v8f;

static constexpr int NB   = 8;     // batch
static constexpr int NS   = 4096;  // sequence
static constexpr int NH   = 16;    // heads
static constexpr int ND   = 64;    // head dim
static constexpr int NDIM = 1024;  // model dim
static constexpr int NG   = 409;   // global tokens  int(4096*0.1)
static constexpr int NL   = 3687;  // local tokens
static constexpr int NM   = NB * NS;
static constexpr float SOFT_SCALE = 0.125f;                   // 64^-0.5
static constexpr float LOG1E4_OVER_32 = 0.2878231366242557f;  // ln(10000)/32

static __device__ __forceinline__ v8f wmma_bf16(v16bf a, v16bf b, v8f c) {
  return __builtin_amdgcn_wmma_f32_16x16x32_bf16(false, a, false, b, (short)0, c,
                                                 false, false);
}

// ---- CDNA5 async global->LDS copy (ASYNCcnt-tracked, CDNA5 ISA) ----
static __device__ __forceinline__ void async_b128(const bf16* g, bf16* l) {
  // Low 32 bits of the generic address of a __shared__ object are the LDS byte address.
  unsigned lds = (unsigned)(unsigned long long)l;
  asm volatile("global_load_async_to_lds_b128 %0, %1, off" ::"v"(lds), "v"(g)
               : "memory");
}
#define WAIT_ASYNC() asm volatile("s_wait_asynccnt 0x0" ::: "memory")

// ---- fragment helpers (layouts per CDNA5 ISA §7.12.2, wave32) ----
// A 16x32 bf16, row-major source: lane m=lane&15, K = (i/8)*16 + (lane/16)*8 + i%8
static __device__ __forceinline__ v16bf frag_a_rm(const bf16* p, int ld) {
  const int lane = threadIdx.x & 31;
  const bf16* r = p + (lane & 15) * ld + ((lane >> 4) << 3);
  v16bf f;
#pragma unroll
  for (int i = 0; i < 8; ++i) { f[i] = r[i]; f[i + 8] = r[16 + i]; }
  return f;
}
// B 32x16 bf16 from [N x K] storage (contiguous): lane n=lane&15, K = (i/8)*16 + (lane/16)*8 + i%8
static __device__ __forceinline__ v16bf frag_b_nk(const bf16* p, int ld) {
  const int lane = threadIdx.x & 31;
  const bf16* r = p + (lane & 15) * ld + ((lane >> 4) << 3);
  v16bf f;
#pragma unroll
  for (int i = 0; i < 8; ++i) { f[i] = r[i]; f[i + 8] = r[16 + i]; }
  return f;
}
// C/D 16x16 f32: lane n=lane&15, rows m = (lane/16)*8 + r
static __device__ __forceinline__ void store_c_f32(float* p, int ld, v8f c) {
  const int lane = threadIdx.x & 31;
  const int n = lane & 15, m0 = (lane >> 4) << 3;
#pragma unroll
  for (int r = 0; r < 8; ++r) p[(size_t)(m0 + r) * ld + n] = c[r];
}
static __device__ __forceinline__ void store_c_bf16(bf16* p, int ld, v8f c) {
  const int lane = threadIdx.x & 31;
  const int n = lane & 15, m0 = (lane >> 4) << 3;
#pragma unroll
  for (int r = 0; r < 8; ++r) p[(size_t)(m0 + r) * ld + n] = (bf16)c[r];
}

// ---------------- conversion kernels ----------------
__global__ void k_cvt_bf16(const float* __restrict__ in, bf16* __restrict__ out, int n) {
  int i = blockIdx.x * blockDim.x + threadIdx.x;
  const int stride = gridDim.x * blockDim.x;
  for (; i < n; i += stride) out[i] = (bf16)in[i];
}
// W [K=1024][N=1024] row-major -> Wt [N][K] bf16
__global__ void k_cvt_w_t(const float* __restrict__ w, bf16* __restrict__ wt) {
  int i = blockIdx.x * blockDim.x + threadIdx.x;
  const int stride = gridDim.x * blockDim.x;
  for (; i < NDIM * NDIM; i += stride) {
    int nn = i >> 10, kk = i & 1023;
    wt[i] = (bf16)w[kk * NDIM + nn];
  }
}

// ---------------- QKV projection + fused RoPE ----------------
// C[64x64] = x[64x1024] * W ; out layout (b, h, n, d) bf16. mode: 0=q(rope) 1=k(rope) 2=v
__global__ __launch_bounds__(128) void k_gemm_qkv(const bf16* __restrict__ xb,
                                                  const bf16* __restrict__ wt,
                                                  bf16* __restrict__ out, int mode) {
  __shared__ bf16 As[2][64][40];
  __shared__ bf16 Bs[2][64][40];
  __shared__ float Cs[64][68];
  const int tid = threadIdx.x;
  const int wid = tid >> 5;
  const int wy = wid >> 1, wx = wid & 1;
  const int m0 = blockIdx.x * 64;  // flat row b*4096+n
  const int n0 = blockIdx.y * 64;  // column = h*64 + d
  v8f c00 = {}, c01 = {}, c10 = {}, c11 = {};
  const int lrow = tid >> 1;
  const int lcol = (tid & 1) << 4;
  const bf16* ag = xb + (size_t)(m0 + lrow) * NDIM + lcol;
  const bf16* bg = wt + (size_t)(n0 + lrow) * NDIM + lcol;

  auto stage = [&](int buf, int k0) {
    async_b128(ag + k0,     &As[buf][lrow][lcol]);
    async_b128(ag + k0 + 8, &As[buf][lrow][lcol + 8]);
    async_b128(bg + k0,     &Bs[buf][lrow][lcol]);
    async_b128(bg + k0 + 8, &Bs[buf][lrow][lcol + 8]);
  };

  stage(0, 0);
  WAIT_ASYNC();
  __syncthreads();
  int buf = 0;
  for (int k0 = 0; k0 < NDIM; k0 += 32) {
    if (k0 + 32 < NDIM) stage(buf ^ 1, k0 + 32);  // overlap fetch with WMMA
    v16bf a0 = frag_a_rm(&As[buf][wy * 32][0], 40);
    v16bf a1 = frag_a_rm(&As[buf][wy * 32 + 16][0], 40);
    v16bf b0 = frag_b_nk(&Bs[buf][wx * 32][0], 40);
    v16bf b1 = frag_b_nk(&Bs[buf][wx * 32 + 16][0], 40);
    c00 = wmma_bf16(a0, b0, c00);
    c01 = wmma_bf16(a0, b1, c01);
    c10 = wmma_bf16(a1, b0, c10);
    c11 = wmma_bf16(a1, b1, c11);
    WAIT_ASYNC();
    __syncthreads();
    buf ^= 1;
  }
  store_c_f32(&Cs[wy * 32][wx * 32], 68, c00);
  store_c_f32(&Cs[wy * 32][wx * 32 + 16], 68, c01);
  store_c_f32(&Cs[wy * 32 + 16][wx * 32], 68, c10);
  store_c_f32(&Cs[wy * 32 + 16][wx * 32 + 16], 68, c11);
  __syncthreads();
  const int h = blockIdx.y;  // 64-col tile == one head
  if (mode < 2) {
    for (int idx = tid; idx < 64 * 32; idx += 128) {
      int row = idx >> 5, j = idx & 31;
      int gr = m0 + row;
      int b = gr >> 12, pos = gr & 4095;
      float t1 = Cs[row][j], t2 = Cs[row][j + 32];
      float ang = (float)pos * __expf(-(float)j * LOG1E4_OVER_32);
      float s, cc;
      __sincosf(ang, &s, &cc);
      bf16* dst = out + (((size_t)(b * NH + h)) * NS + pos) * ND;
      dst[j]      = (bf16)(t1 * cc - t2 * s);
      dst[j + 32] = (bf16)(t1 * s + t2 * cc);
    }
  } else {
    for (int idx = tid; idx < 64 * 64; idx += 128) {
      int row = idx >> 6, d = idx & 63;
      int gr = m0 + row;
      int b = gr >> 12, pos = gr & 4095;
      out[(((size_t)(b * NH + h)) * NS + pos) * ND + d] = (bf16)Cs[row][d];
    }
  }
}

// ---------------- global attention (flash-style, first NG tokens) ----------------
__global__ __launch_bounds__(128) void k_global_attn(const bf16* __restrict__ qb,
                                                     const bf16* __restrict__ kb,
                                                     const bf16* __restrict__ vb,
                                                     bf16* __restrict__ cb) {
  __shared__ bf16 Ks[32][72];  // [j][d]   (B for scores: [N=j x K=d], contiguous)
  __shared__ bf16 Vs[64][40];  // [d][j]   transposed (B for P*V: [N=d x K=j], contiguous)
  __shared__ bf16 Ps[4][16][40];
  const int tid = threadIdx.x;
  const int w = tid >> 5;
  const int lane = tid & 31;
  const int h = blockIdx.y, b = blockIdx.z;
  const int bh = b * NH + h;
  const int i0 = blockIdx.x * 64 + w * 16;  // query-row base for this wave
  const bf16* qbase = qb + ((size_t)bh * NS + i0) * ND;
  const bf16* kbase = kb + (size_t)bh * NS * ND;
  const bf16* vbase = vb + (size_t)bh * NS * ND;

  v16bf aq0 = frag_a_rm(qbase, ND);       // d 0..31
  v16bf aq1 = frag_a_rm(qbase + 32, ND);  // d 32..63

  v8f o0 = {}, o1 = {}, o2 = {}, o3 = {};
  float mrun[8], lrun[8];
#pragma unroll
  for (int r = 0; r < 8; ++r) { mrun[r] = -1e30f; lrun[r] = 0.f; }
  const int n = lane & 15;
  const int m0 = (lane >> 4) << 3;

  for (int j0 = 0; j0 < NG; j0 += 32) {
    {  // stage K (row-major) and V (transposed); 128 chunks of 16 = one per thread
      int r = tid >> 2, c0 = (tid & 3) << 4;
      int j = j0 + r;
      if (j < NG) {
        const bf16* ksrc = kbase + (size_t)j * ND + c0;
        *(uint4*)&Ks[r][c0]     = *(const uint4*)ksrc;
        *(uint4*)&Ks[r][c0 + 8] = *(const uint4*)(ksrc + 8);
        const bf16* vsrc = vbase + (size_t)j * ND + c0;
        bf16 tmp[16];
        *(uint4*)&tmp[0] = *(const uint4*)vsrc;
        *(uint4*)&tmp[8] = *(const uint4*)(vsrc + 8);
#pragma unroll
        for (int t = 0; t < 16; ++t) Vs[c0 + t][r] = tmp[t];
        if (j0 + 32 < NG) __builtin_prefetch(ksrc + 32 * ND, 0, 0);
      } else {
        uint4 z = {0, 0, 0, 0};
        *(uint4*)&Ks[r][c0] = z;
        *(uint4*)&Ks[r][c0 + 8] = z;
#pragma unroll
        for (int t = 0; t < 16; ++t) Vs[c0 + t][r] = (bf16)0.f;
      }
    }
    __syncthreads();
    v8f s0 = {}, s1 = {};
    s0 = wmma_bf16(aq0, frag_b_nk(&Ks[0][0], 72), s0);
    s0 = wmma_bf16(aq1, frag_b_nk(&Ks[0][32], 72), s0);
    s1 = wmma_bf16(aq0, frag_b_nk(&Ks[16][0], 72), s1);
    s1 = wmma_bf16(aq1, frag_b_nk(&Ks[16][32], 72), s1);
    const bool v0 = (j0 + n) < NG;
    const bool v1 = (j0 + 16 + n) < NG;
    float p0[8], p1[8];
#pragma unroll
    for (int r = 0; r < 8; ++r) {
      float x0 = v0 ? s0[r] * SOFT_SCALE : -1e30f;
      float x1 = v1 ? s1[r] * SOFT_SCALE : -1e30f;
      float mx = fmaxf(x0, x1);
      mx = fmaxf(mx, __shfl_xor(mx, 1, 32));
      mx = fmaxf(mx, __shfl_xor(mx, 2, 32));
      mx = fmaxf(mx, __shfl_xor(mx, 4, 32));
      mx = fmaxf(mx, __shfl_xor(mx, 8, 32));
      float newm = fmaxf(mrun[r], mx);
      float sc = __expf(mrun[r] - newm);
      float e0 = __expf(x0 - newm);
      float e1 = __expf(x1 - newm);
      float ps = e0 + e1;
      ps += __shfl_xor(ps, 1, 32);
      ps += __shfl_xor(ps, 2, 32);
      ps += __shfl_xor(ps, 4, 32);
      ps += __shfl_xor(ps, 8, 32);
      lrun[r] = lrun[r] * sc + ps;
      mrun[r] = newm;
      o0[r] *= sc; o1[r] *= sc; o2[r] *= sc; o3[r] *= sc;
      p0[r] = e0; p1[r] = e1;
    }
#pragma unroll
    for (int r = 0; r < 8; ++r) {  // C-layout -> A-layout transpose via LDS
      Ps[w][m0 + r][n]      = (bf16)p0[r];
      Ps[w][m0 + r][n + 16] = (bf16)p1[r];
    }
    v16bf ap = frag_a_rm(&Ps[w][0][0], 40);
    o0 = wmma_bf16(ap, frag_b_nk(&Vs[0][0], 40), o0);
    o1 = wmma_bf16(ap, frag_b_nk(&Vs[16][0], 40), o1);
    o2 = wmma_bf16(ap, frag_b_nk(&Vs[32][0], 40), o2);
    o3 = wmma_bf16(ap, frag_b_nk(&Vs[48][0], 40), o3);
    __syncthreads();
  }
#pragma unroll
  for (int r = 0; r < 8; ++r) {
    int tok = i0 + m0 + r;
    if (tok >= NG) continue;
    float inv = 1.f / lrun[r];
    bf16* dst = cb + ((size_t)b * NS + tok) * NDIM + h * ND;
    dst[n]      = (bf16)(o0[r] * inv);
    dst[n + 16] = (bf16)(o1[r] * inv);
    dst[n + 32] = (bf16)(o2[r] * inv);
    dst[n + 48] = (bf16)(o3[r] * inv);
  }
}

// ---------------- softmax of k_local over sequence (per b,h,d column) ----------------
__global__ __launch_bounds__(256) void k_softmax_kl(const bf16* __restrict__ kb,
                                                    bf16* __restrict__ ks) {
  const int bh = blockIdx.x;
  const int tid = threadIdx.x;
  const int d = tid & 63, grp = tid >> 6;
  __shared__ float red[4][64];
  const bf16* src = kb + ((size_t)bh * NS + NG) * ND;
  bf16* dst = ks + (size_t)bh * NL * ND;
  float mx = -1e30f;
  for (int r = grp; r < NL; r += 4) mx = fmaxf(mx, (float)src[(size_t)r * ND + d]);
  red[grp][d] = mx;
  __syncthreads();
  float cmax = fmaxf(fmaxf(red[0][d], red[1][d]), fmaxf(red[2][d], red[3][d]));
  __syncthreads();
  float sm = 0.f;
  for (int r = grp; r < NL; r += 4) sm += __expf((float)src[(size_t)r * ND + d] - cmax);
  red[grp][d] = sm;
  __syncthreads();
  float inv = 1.f / (red[0][d] + red[1][d] + red[2][d] + red[3][d]);
  for (int r = grp; r < NL; r += 4)
    dst[(size_t)r * ND + d] = (bf16)(__expf((float)src[(size_t)r * ND + d] - cmax) * inv);
}

// ---------------- kv[b,h,d,e] = sum_l v[l,d] * k_sm[l,e]  (K=3687 GEMM) ----------------
__global__ __launch_bounds__(128) void k_kv(const bf16* __restrict__ vb,
                                            const bf16* __restrict__ ks,
                                            bf16* __restrict__ kvb) {
  __shared__ bf16 Vt[64][40];  // [d][l] transposed -> A rows contiguous
  __shared__ bf16 Kt[64][40];  // [e][l] transposed -> B [N x K] contiguous
  const int tid = threadIdx.x;
  const int w = tid >> 5;
  const int bh = blockIdx.x;
  const bf16* vbase = vb + ((size_t)bh * NS + NG) * ND;
  const bf16* kbase = ks + (size_t)bh * NL * ND;
  v8f c0 = {}, c1 = {}, c2 = {}, c3 = {};
  for (int l0 = 0; l0 < NL; l0 += 32) {
    {  // transpose-stage both tiles; one 16-wide chunk per thread
      int r = tid >> 2, c0i = (tid & 3) << 4;
      int l = l0 + r;
      bf16 tv[16], tk[16];
      if (l < NL) {
        const bf16* vs = vbase + (size_t)l * ND + c0i;
        const bf16* kss = kbase + (size_t)l * ND + c0i;
        *(uint4*)&tv[0] = *(const uint4*)vs;
        *(uint4*)&tv[8] = *(const uint4*)(vs + 8);
        *(uint4*)&tk[0] = *(const uint4*)kss;
        *(uint4*)&tk[8] = *(const uint4*)(kss + 8);
        if (l0 + 32 < NL) {
          __builtin_prefetch(vs + 32 * ND, 0, 0);
          __builtin_prefetch(kss + 32 * ND, 0, 0);
        }
      } else {
#pragma unroll
        for (int t = 0; t < 16; ++t) { tv[t] = (bf16)0.f; tk[t] = (bf16)0.f; }
      }
#pragma unroll
      for (int t = 0; t < 16; ++t) { Vt[c0i + t][r] = tv[t]; Kt[c0i + t][r] = tk[t]; }
    }
    __syncthreads();
    v16bf a = frag_a_rm(&Vt[w * 16][0], 40);  // A[m=d][k=l]
    c0 = wmma_bf16(a, frag_b_nk(&Kt[0][0], 40), c0);
    c1 = wmma_bf16(a, frag_b_nk(&Kt[16][0], 40), c1);
    c2 = wmma_bf16(a, frag_b_nk(&Kt[32][0], 40), c2);
    c3 = wmma_bf16(a, frag_b_nk(&Kt[48][0], 40), c3);
    __syncthreads();
  }
  bf16* dst = kvb + (size_t)bh * ND * ND + (size_t)(w * 16) * ND;
  store_c_bf16(dst, ND, c0);
  store_c_bf16(dst + 16, ND, c1);
  store_c_bf16(dst + 32, ND, c2);
  store_c_bf16(dst + 48, ND, c3);
}

// ---------------- local_out[l,e] = sum_d q[l,d] * kv[d,e] ----------------
__global__ __launch_bounds__(128) void k_local_out(const bf16* __restrict__ qb,
                                                   const bf16* __restrict__ kvb,
                                                   bf16* __restrict__ cb) {
  __shared__ bf16 KVt[64][72];  // [e][d] transposed -> B [N x K] contiguous
  __shared__ bf16 Qs[64][72];   // [l][d] row-major
  const int tid = threadIdx.x;
  const int w = tid >> 5;
  const int h = blockIdx.y, b = blockIdx.z;
  const int bh = b * NH + h;
  const int l0 = blockIdx.x * 64;
  const bf16* kvbase = kvb + (size_t)bh * ND * ND;
  const bf16* qbase = qb + ((size_t)bh * NS + NG) * ND;
  for (int idx = tid; idx < 64 * 4; idx += 128) {
    int r = idx >> 2, c0 = (idx & 3) << 4;
    bf16 tmp[16];
    *(uint4*)&tmp[0] = *(const uint4*)(kvbase + (size_t)r * ND + c0);
    *(uint4*)&tmp[8] = *(const uint4*)(kvbase + (size_t)r * ND + c0 + 8);
#pragma unroll
    for (int t = 0; t < 16; ++t) KVt[c0 + t][r] = tmp[t];
    int l = l0 + r;
    if (l < NL) {
      const bf16* qs = qbase + (size_t)l * ND + c0;
      *(uint4*)&Qs[r][c0]     = *(const uint4*)qs;
      *(uint4*)&Qs[r][c0 + 8] = *(const uint4*)(qs + 8);
    } else {
      uint4 z = {0, 0, 0, 0};
      *(uint4*)&Qs[r][c0] = z;
      *(uint4*)&Qs[r][c0 + 8] = z;
    }
  }
  __syncthreads();
  v8f c0 = {}, c1 = {}, c2 = {}, c3 = {};
#pragma unroll
  for (int d0 = 0; d0 < 64; d0 += 32) {
    v16bf a = frag_a_rm(&Qs[w * 16][d0], 72);
    c0 = wmma_bf16(a, frag_b_nk(&KVt[0][d0], 72), c0);
    c1 = wmma_bf16(a, frag_b_nk(&KVt[16][d0], 72), c1);
    c2 = wmma_bf16(a, frag_b_nk(&KVt[32][d0], 72), c2);
    c3 = wmma_bf16(a, frag_b_nk(&KVt[48][d0], 72), c3);
  }
  const int lane = tid & 31;
  const int nn = lane & 15, m0 = (lane >> 4) << 3;
#pragma unroll
  for (int r = 0; r < 8; ++r) {
    int l = l0 + w * 16 + m0 + r;
    if (l >= NL) continue;
    bf16* dst = cb + ((size_t)b * NS + NG + l) * NDIM + h * ND;
    dst[nn]      = (bf16)c0[r];
    dst[nn + 16] = (bf16)c1[r];
    dst[nn + 32] = (bf16)c2[r];
    dst[nn + 48] = (bf16)c3[r];
  }
}

// ---------------- output projection: out = concat @ Wout (f32 result) ----------------
__global__ __launch_bounds__(128) void k_gemm_out(const bf16* __restrict__ cb,
                                                  const bf16* __restrict__ wt,
                                                  float* __restrict__ out) {
  __shared__ bf16 As[2][64][40];
  __shared__ bf16 Bs[2][64][40];
  const int tid = threadIdx.x;
  const int wid = tid >> 5;
  const int wy = wid >> 1, wx = wid & 1;
  const int m0 = blockIdx.x * 64;
  const int n0 = blockIdx.y * 64;
  v8f c00 = {}, c01 = {}, c10 = {}, c11 = {};
  const int lrow = tid >> 1;
  const int lcol = (tid & 1) << 4;
  const bf16* ag = cb + (size_t)(m0 + lrow) * NDIM + lcol;
  const bf16* bg = wt + (size_t)(n0 + lrow) * NDIM + lcol;

  auto stage = [&](int buf, int k0) {
    async_b128(ag + k0,     &As[buf][lrow][lcol]);
    async_b128(ag + k0 + 8, &As[buf][lrow][lcol + 8]);
    async_b128(bg + k0,     &Bs[buf][lrow][lcol]);
    async_b128(bg + k0 + 8, &Bs[buf][lrow][lcol + 8]);
  };

  stage(0, 0);
  WAIT_ASYNC();
  __syncthreads();
  int buf = 0;
  for (int k0 = 0; k0 < NDIM; k0 += 32) {
    if (k0 + 32 < NDIM) stage(buf ^ 1, k0 + 32);
    v16bf a0 = frag_a_rm(&As[buf][wy * 32][0], 40);
    v16bf a1 = frag_a_rm(&As[buf][wy * 32 + 16][0], 40);
    v16bf b0 = frag_b_nk(&Bs[buf][wx * 32][0], 40);
    v16bf b1 = frag_b_nk(&Bs[buf][wx * 32 + 16][0], 40);
    c00 = wmma_bf16(a0, b0, c00);
    c01 = wmma_bf16(a0, b1, c01);
    c10 = wmma_bf16(a1, b0, c10);
    c11 = wmma_bf16(a1, b1, c11);
    WAIT_ASYNC();
    __syncthreads();
    buf ^= 1;
  }
  float* dst = out + (size_t)m0 * NDIM + n0;
  store_c_f32(dst + (size_t)(wy * 32) * NDIM + wx * 32, NDIM, c00);
  store_c_f32(dst + (size_t)(wy * 32) * NDIM + wx * 32 + 16, NDIM, c01);
  store_c_f32(dst + (size_t)(wy * 32 + 16) * NDIM + wx * 32, NDIM, c10);
  store_c_f32(dst + (size_t)(wy * 32 + 16) * NDIM + wx * 32 + 16, NDIM, c11);
}

extern "C" void kernel_launch(void* const* d_in, const int* in_sizes, int n_in,
                              void* d_out, int out_size, void* d_ws, size_t ws_size,
                              hipStream_t stream) {
  (void)in_sizes; (void)n_in; (void)out_size; (void)ws_size;
  const float* x    = (const float*)d_in[0];
  const float* Wq   = (const float*)d_in[1];
  const float* Wk   = (const float*)d_in[2];
  const float* Wv   = (const float*)d_in[3];
  const float* Wout = (const float*)d_in[4];

  char* wsp = (char*)d_ws;
  auto alloc = [&](size_t elems) -> bf16* {
    bf16* p = (bf16*)wsp;
    wsp += (elems * sizeof(bf16) + 255) & ~(size_t)255;
    return p;
  };
  bf16* xb  = alloc((size_t)NM * NDIM);
  bf16* wqt = alloc((size_t)NDIM * NDIM);
  bf16* wkt = alloc((size_t)NDIM * NDIM);
  bf16* wvt = alloc((size_t)NDIM * NDIM);
  bf16* wot = alloc((size_t)NDIM * NDIM);
  bf16* qb  = alloc((size_t)NB * NH * NS * ND);
  bf16* kb  = alloc((size_t)NB * NH * NS * ND);
  bf16* vb  = alloc((size_t)NB * NH * NS * ND);
  bf16* ks  = alloc((size_t)NB * NH * NL * ND);
  bf16* kvb = alloc((size_t)NB * NH * ND * ND);
  bf16* cb  = alloc((size_t)NM * NDIM);

  k_cvt_bf16<<<2048, 256, 0, stream>>>(x, xb, NM * NDIM);
  k_cvt_w_t<<<512, 256, 0, stream>>>(Wq, wqt);
  k_cvt_w_t<<<512, 256, 0, stream>>>(Wk, wkt);
  k_cvt_w_t<<<512, 256, 0, stream>>>(Wv, wvt);
  k_cvt_w_t<<<512, 256, 0, stream>>>(Wout, wot);

  dim3 gq(NM / 64, NDIM / 64);
  k_gemm_qkv<<<gq, 128, 0, stream>>>(xb, wqt, qb, 0);
  k_gemm_qkv<<<gq, 128, 0, stream>>>(xb, wkt, kb, 1);
  k_gemm_qkv<<<gq, 128, 0, stream>>>(xb, wvt, vb, 2);

  dim3 ga((NG + 63) / 64, NH, NB);
  k_global_attn<<<ga, 128, 0, stream>>>(qb, kb, vb, cb);

  k_softmax_kl<<<NB * NH, 256, 0, stream>>>(kb, ks);
  k_kv<<<NB * NH, 128, 0, stream>>>(vb, ks, kvb);

  dim3 gl((NL + 63) / 64, NH, NB);
  k_local_out<<<gl, 128, 0, stream>>>(qb, kvb, cb);

  dim3 go(NM / 64, NDIM / 64);
  k_gemm_out<<<go, 128, 0, stream>>>(cb, wot, (float*)d_out);
}
